// MarketScoringUnit_3152505995659
// MI455X (gfx1250) — matrix operand order repaired and checked by reading, not compiled, not run on hardware
//
#include <hip/hip_runtime.h>
#include <hip/hip_bf16.h>
#include <math.h>

// Problem dims (fixed by the reference)
#define NB 128   // batch
#define NT 512   // time steps
#define NH 512   // hidden
#define NG 2048  // 4*NH gate width
#define NKT (NH / 32)   // 16 K-tiles per K=512 GEMM
#define LSTM_WGS 32     // persistent grid for the recurrent phase

typedef __attribute__((ext_vector_type(16))) __bf16 v16bf;
typedef __attribute__((ext_vector_type(8)))  __bf16 v8bf;
typedef __attribute__((ext_vector_type(8)))  float  v8f;

static __device__ __forceinline__ float bf2f(__bf16 b) {
  unsigned short h = __builtin_bit_cast(unsigned short, b);
  unsigned u = ((unsigned)h) << 16;
  return __builtin_bit_cast(float, u);
}
static __device__ __forceinline__ __bf16 f2bf(float f) {
  unsigned u = __builtin_bit_cast(unsigned, f);
  unsigned r = u + 0x7fffu + ((u >> 16) & 1u);   // round-to-nearest-even
  unsigned short h = (unsigned short)(r >> 16);
  return __builtin_bit_cast(__bf16, h);
}
static __device__ __forceinline__ float sigm(float x) { return 1.0f / (1.0f + __expf(-x)); }

// A-fragment in this code's contiguous k order (same order as B; not AMD's table order):
// lanes 0-15 hold row M=lane, K={0..15}; lanes 16-31 hold K={16..31}.
// Caller passes p = rowBase + k0 + (lane>=16 ? 16 : 0): two contiguous 16B loads.
static __device__ __forceinline__ v16bf load_a_frag(const __bf16* p) {
  v8bf lo = *(const v8bf*)(p);
  v8bf hh = *(const v8bf*)(p + 8);
  return __builtin_shufflevector(lo, hh, 0,1,2,3,4,5,6,7,8,9,10,11,12,13,14,15);
}
// B-fragment (32x16 KxN): lane n holds column n, 16 consecutive K values
// (K base 0 for lanes 0-15, 16 for lanes 16-31) -> one contiguous 32B load from
// a (N,K) row-major weight (the reference's (out,in) layout, no transpose).

#define WMMA_BF16(A, Bm, C) \
  __builtin_amdgcn_wmma_f32_16x16x32_bf16(false, (A), false, (Bm), (short)0, (C), false, false)

// xg is stored in D-fragment tile order so the serial LSTM phase reads it with
// vector loads: tile (t, bt, ng, j) -> 256 bf16, element offset lane*8 + r.
static __device__ __forceinline__ size_t xg_tile(int t, int bt, int ng, int j) {
  return ((((size_t)t * 8 + bt) * 32 + ng) * 4 + j) * 256;
}

// ---------------------------------------------------------------------------
// K0: convert operands to bf16, combine biases, zero h0 + barrier counter.
// ---------------------------------------------------------------------------
__global__ void k_prep(const float* __restrict__ x, const float* __restrict__ Wih,
                       const float* __restrict__ Whh, const float* __restrict__ bih,
                       const float* __restrict__ bhh, const float* __restrict__ U1,
                       const float* __restrict__ U2,
                       __bf16* x_bf, __bf16* wih_bf, __bf16* whh_bf,
                       __bf16* u1_bf, __bf16* u2_bf, float* bias,
                       __bf16* hA, int* bar) {
  size_t i = (size_t)blockIdx.x * blockDim.x + threadIdx.x;
  size_t stride = (size_t)gridDim.x * blockDim.x;
  const size_t NX = (size_t)NB * NT * NH;
  for (size_t k = i; k < NX; k += stride) {
    x_bf[k] = f2bf(x[k]);
    if (k < (size_t)NG * NH) { wih_bf[k] = f2bf(Wih[k]); whh_bf[k] = f2bf(Whh[k]); }
    if (k < (size_t)NH * 2 * NH) u1_bf[k] = f2bf(U1[k]);
    if (k < (size_t)NH * NH)     u2_bf[k] = f2bf(U2[k]);
    if (k < NG)                  bias[k]  = bih[k] + bhh[k];
    if (k < (size_t)NB * NH)     hA[k]    = f2bf(0.f);
    if (k < 64)                  bar[k]   = 0;
  }
}

// ---------------------------------------------------------------------------
// K1: xg = x·W_ihᵀ + bias, stored bf16 in D-fragment tile order.
// One wave per (t, 16-row b tile, 64-col n group). Fully-unrolled ping/pong
// K pipeline; launch_bounds(256,2) widens the VGPR budget so several fragment
// sets stay live and loads run ahead of the WMMAs.
// ---------------------------------------------------------------------------
__global__ void __launch_bounds__(256, 2) k_xg(const __bf16* __restrict__ x_bf,
                                               const __bf16* __restrict__ wih_bf,
                                               const float* __restrict__ bias,
                                               __bf16* __restrict__ xg) {
  const int lane = threadIdx.x & 31, hi = lane >> 4, ln = lane & 15;
  const int id = blockIdx.x * 8 + (threadIdx.x >> 5);
  const int ng = id & 31, rest = id >> 5;
  const int bt = rest & 7, t = rest >> 3;
  const int n0 = ng * 64;
  const __bf16* arow = x_bf + ((size_t)(bt * 16 + ln) * NT + t) * NH + hi * 16;
  const __bf16* brow[4];
  for (int j = 0; j < 4; ++j)
    brow[j] = wih_bf + (size_t)(n0 + j * 16 + ln) * NH + hi * 16;

  v8f acc[4];
  for (int j = 0; j < 4; ++j) {
    float bv = bias[n0 + j * 16 + ln];
    acc[j] = (v8f){bv, bv, bv, bv, bv, bv, bv, bv};
  }
  v16bf aP = load_a_frag(arow);
  v16bf bP[4];
  for (int j = 0; j < 4; ++j) bP[j] = *(const v16bf*)(brow[j]);
#pragma unroll
  for (int kt = 0; kt < NKT - 2; kt += 2) {
    v16bf aQ = load_a_frag(arow + (kt + 1) * 32);
    v16bf bQ[4];
    for (int j = 0; j < 4; ++j) bQ[j] = *(const v16bf*)(brow[j] + (kt + 1) * 32);
    for (int j = 0; j < 4; ++j) acc[j] = WMMA_BF16(aP, bP[j], acc[j]);
    aP = load_a_frag(arow + (kt + 2) * 32);
    for (int j = 0; j < 4; ++j) bP[j] = *(const v16bf*)(brow[j] + (kt + 2) * 32);
    for (int j = 0; j < 4; ++j) acc[j] = WMMA_BF16(aQ, bQ[j], acc[j]);
  }
  {  // epilogue: tiles NKT-2 (in P) and NKT-1
    v16bf aQ = load_a_frag(arow + (NKT - 1) * 32);
    v16bf bQ[4];
    for (int j = 0; j < 4; ++j) bQ[j] = *(const v16bf*)(brow[j] + (NKT - 1) * 32);
    for (int j = 0; j < 4; ++j) acc[j] = WMMA_BF16(aP, bP[j], acc[j]);
    for (int j = 0; j < 4; ++j) acc[j] = WMMA_BF16(aQ, bQ[j], acc[j]);
  }
  // store each 16x16 tile in D-fragment order: one 16B store per tile per lane
  for (int j = 0; j < 4; ++j) {
    v8bf pk;
    for (int r = 0; r < 8; ++r) pk[r] = f2bf(acc[j][r]);
    *(v8bf*)(xg + xg_tile(t, bt, ng, j) + lane * 8) = pk;
  }
}

// ---------------------------------------------------------------------------
// K2: persistent recurrent LSTM. 32 WGs x 8 waves. WG jt owns h cols
// [16jt,16jt+16) => gate tiles {jt, jt+32, jt+64, jt+96} of the 2048-wide GEMM.
// W_hh slice (64x512 bf16 = 64KB) lives in LDS; c state lives in acc registers.
// Exactly 1 block/WGP = 2 waves/SIMD, so launch_bounds(256,2) states the true
// occupancy and unlocks the VGPR file for deep fragment pipelining.
// ---------------------------------------------------------------------------
__global__ void __launch_bounds__(256, 2) k_lstm(const __bf16* __restrict__ xg,
                                                 const __bf16* __restrict__ whh_bf,
                                                 __bf16* hA, __bf16* hB,
                                                 __bf16* __restrict__ hs, int* bar) {
  __shared__ __align__(32) __bf16 lds_w[64 * NH];
  const int jt = blockIdx.x;
  const int tid = threadIdx.x;
  const int lane = tid & 31, ww = tid >> 5, hi = lane >> 4, ln = lane & 15;

  // Stage this WG's W_hh rows {gate*512 + jt*16 + rr} into LDS, once.
  for (int idx = tid; idx < 64 * NH / 8; idx += 256) {
    int row = idx >> 6, c8 = (idx & 63) * 8;
    int gate = row >> 4, rr = row & 15;
    *(v8bf*)&lds_w[row * NH + c8] =
        *(const v8bf*)(whh_bf + (size_t)(gate * NH + jt * 16 + rr) * NH + c8);
  }
  __syncthreads();

  const int m0 = ww * 16;           // wave's 16 batch rows
  const int colh = jt * 16 + ln;    // wave/lane h column
  const int ngbase = jt >> 2, jsub = jt & 3;
  const __bf16* lrow[4];
  for (int j = 0; j < 4; ++j) lrow[j] = &lds_w[(j * 16 + ln) * NH + hi * 16];
  v8f c = (v8f){0, 0, 0, 0, 0, 0, 0, 0};

  for (int t = 0; t < NT; ++t) {
    const __bf16* hin = (t & 1) ? hB : hA;
    __bf16* hout = (t & 1) ? hA : hB;

    // accumulators initialized from xg tiles: 1 vector load per gate per lane
    v8f acc[4];
    for (int j = 0; j < 4; ++j) {
      v8bf pk = *(const v8bf*)(xg + xg_tile(t, ww, j * 8 + ngbase, jsub) + lane * 8);
      v8f a;
      for (int r = 0; r < 8; ++r) a[r] = bf2f(pk[r]);
      acc[j] = a;
    }
    if (t + 1 < NT)  // hint next step's xg tiles toward the caches
      __builtin_prefetch(xg + xg_tile(t + 1, ww, ngbase, jsub) + lane * 8, 0, 1);

    // fully-unrolled ping/pong K pipeline: A from global (L2-hot h), B from LDS
    const __bf16* arow = hin + (size_t)(m0 + ln) * NH + hi * 16;
    v16bf aP = load_a_frag(arow);
    v16bf bP[4];
    for (int j = 0; j < 4; ++j) bP[j] = *(const v16bf*)(lrow[j]);
#pragma unroll
    for (int kt = 0; kt < NKT - 2; kt += 2) {
      v16bf aQ = load_a_frag(arow + (kt + 1) * 32);
      v16bf bQ[4];
      for (int j = 0; j < 4; ++j) bQ[j] = *(const v16bf*)(lrow[j] + (kt + 1) * 32);
      for (int j = 0; j < 4; ++j) acc[j] = WMMA_BF16(aP, bP[j], acc[j]);
      aP = load_a_frag(arow + (kt + 2) * 32);
      for (int j = 0; j < 4; ++j) bP[j] = *(const v16bf*)(lrow[j] + (kt + 2) * 32);
      for (int j = 0; j < 4; ++j) acc[j] = WMMA_BF16(aQ, bQ[j], acc[j]);
    }
    {  // epilogue: tiles NKT-2 (in P) and NKT-1
      v16bf aQ = load_a_frag(arow + (NKT - 1) * 32);
      v16bf bQ[4];
      for (int j = 0; j < 4; ++j) bQ[j] = *(const v16bf*)(lrow[j] + (NKT - 1) * 32);
      for (int j = 0; j < 4; ++j) acc[j] = WMMA_BF16(aP, bP[j], acc[j]);
      for (int j = 0; j < 4; ++j) acc[j] = WMMA_BF16(aQ, bQ[j], acc[j]);
    }
    // Gate math: each wave owns its full 16x16 (b,h) block -> no cross-wave traffic.
    for (int r = 0; r < 8; ++r) {
      float ig = sigm(acc[0][r]);
      float fg = sigm(acc[1][r]);
      float gg = tanhf(acc[2][r]);
      float og = sigm(acc[3][r]);
      float cn = fg * c[r] + ig * gg;
      c[r] = cn;
      __bf16 hb = f2bf(og * tanhf(cn));
      int row = m0 + hi * 8 + r;
      hout[(size_t)row * NH + colh] = hb;                 // next-step GEMM operand
      hs[((size_t)row * NT + t) * NH + colh] = hb;        // history for attention
    }
    // device-wide step barrier (monotone counter; zeroed by k_prep each launch)
    __syncthreads();
    if (tid == 0) {
      __threadfence();
      __hip_atomic_fetch_add(bar, 1, __ATOMIC_RELEASE, __HIP_MEMORY_SCOPE_AGENT);
      const int target = (t + 1) * LSTM_WGS;
      while (__hip_atomic_load(bar, __ATOMIC_ACQUIRE, __HIP_MEMORY_SCOPE_AGENT) < target)
        __builtin_amdgcn_s_sleep(2);
    }
    __syncthreads();
    __threadfence();
  }
}

// ---------------------------------------------------------------------------
// K3: rbuf[b,g] = h_last[b,:]·U1[g, H:2H]   (128x512, K=512)
// ---------------------------------------------------------------------------
__global__ void __launch_bounds__(256, 2) k_hlast(const __bf16* __restrict__ hs,
                                                  const __bf16* __restrict__ u1_bf,
                                                  float* __restrict__ rbuf) {
  const int lane = threadIdx.x & 31, hi = lane >> 4, ln = lane & 15;
  const int id = blockIdx.x * 8 + (threadIdx.x >> 5);
  const int mt = id & 7, nt = id >> 3;
  v8f acc = (v8f){0, 0, 0, 0, 0, 0, 0, 0};
  const __bf16* arow = hs + ((size_t)(mt * 16 + ln) * NT + (NT - 1)) * NH + hi * 16;
  const __bf16* brow = u1_bf + (size_t)(nt * 16 + ln) * (2 * NH) + NH + hi * 16;
  v16bf aP = load_a_frag(arow);
  v16bf bP = *(const v16bf*)(brow);
#pragma unroll
  for (int kt = 0; kt < NKT - 2; kt += 2) {
    v16bf aQ = load_a_frag(arow + (kt + 1) * 32);
    v16bf bQ = *(const v16bf*)(brow + (kt + 1) * 32);
    acc = WMMA_BF16(aP, bP, acc);
    aP = load_a_frag(arow + (kt + 2) * 32);
    bP = *(const v16bf*)(brow + (kt + 2) * 32);
    acc = WMMA_BF16(aQ, bQ, acc);
  }
  {
    v16bf aQ = load_a_frag(arow + (NKT - 1) * 32);
    v16bf bQ = *(const v16bf*)(brow + (NKT - 1) * 32);
    acc = WMMA_BF16(aP, bP, acc);
    acc = WMMA_BF16(aQ, bQ, acc);
  }
  for (int r = 0; r < 8; ++r)
    rbuf[(size_t)(mt * 16 + hi * 8 + r) * NH + nt * 16 + ln] = acc[r];
}

// ---------------------------------------------------------------------------
// K4: fused attention energies:
//   e[b,t] = Ve · tanh( h[b,t]·U1aᵀ + x[b,t]·U2ᵀ + rbuf[b] )
// n tiled in groups of 4 (A-fragments reused 4x); two interleaved WMMA GEMMs
// per accumulator; A streams pipelined with peeled epilogue; tanh + Ve-dot on
// registers with cross-lane shuffle reduction.
// ---------------------------------------------------------------------------
__global__ void __launch_bounds__(256, 2) k_score(const __bf16* __restrict__ hs,
                                                  const __bf16* __restrict__ x_bf,
                                                  const __bf16* __restrict__ u1_bf,
                                                  const __bf16* __restrict__ u2_bf,
                                                  const float* __restrict__ rbuf,
                                                  const float* __restrict__ Ve,
                                                  float* __restrict__ e) {
  const int lane = threadIdx.x & 31, hi = lane >> 4, ln = lane & 15;
  const int id = blockIdx.x * 8 + (threadIdx.x >> 5);
  const int b = id >> 5, tt = id & 31;
  const int trow = tt * 16 + ln;
  const __bf16* ah = hs + ((size_t)b * NT + trow) * NH + hi * 16;
  const __bf16* ax = x_bf + ((size_t)b * NT + trow) * NH + hi * 16;

  float ep[8] = {0, 0, 0, 0, 0, 0, 0, 0};
  for (int ntg = 0; ntg < NH / 64; ++ntg) {       // 8 groups of 4 n-tiles
    const __bf16* b1row[4];
    const __bf16* b2row[4];
    for (int jj = 0; jj < 4; ++jj) {
      const int g = (ntg * 4 + jj) * 16 + ln;
      b1row[jj] = u1_bf + (size_t)g * (2 * NH) + hi * 16;
      b2row[jj] = u2_bf + (size_t)g * NH + hi * 16;
    }
    v8f acc[4];
    for (int jj = 0; jj < 4; ++jj) {
      float rv = rbuf[(size_t)b * NH + (ntg * 4 + jj) * 16 + ln];
      acc[jj] = (v8f){rv, rv, rv, rv, rv, rv, rv, rv};
    }
    v16bf hP = load_a_frag(ah);
    v16bf xP = load_a_frag(ax);
#pragma unroll
    for (int kt = 0; kt < NKT - 2; kt += 2) {
      v16bf hQ = load_a_frag(ah + (kt + 1) * 32);
      v16bf xQ = load_a_frag(ax + (kt + 1) * 32);
      for (int jj = 0; jj < 4; ++jj) {
        acc[jj] = WMMA_BF16(hP, *(const v16bf*)(b1row[jj] + kt * 32), acc[jj]);
        acc[jj] = WMMA_BF16(xP, *(const v16bf*)(b2row[jj] + kt * 32), acc[jj]);
      }
      hP = load_a_frag(ah + (kt + 2) * 32);
      xP = load_a_frag(ax + (kt + 2) * 32);
      for (int jj = 0; jj < 4; ++jj) {
        acc[jj] = WMMA_BF16(hQ, *(const v16bf*)(b1row[jj] + (kt + 1) * 32), acc[jj]);
        acc[jj] = WMMA_BF16(xQ, *(const v16bf*)(b2row[jj] + (kt + 1) * 32), acc[jj]);
      }
    }
    {  // epilogue: tiles NKT-2 (in P) and NKT-1
      v16bf hQ = load_a_frag(ah + (NKT - 1) * 32);
      v16bf xQ = load_a_frag(ax + (NKT - 1) * 32);
      for (int jj = 0; jj < 4; ++jj) {
        acc[jj] = WMMA_BF16(hP, *(const v16bf*)(b1row[jj] + (NKT - 2) * 32), acc[jj]);
        acc[jj] = WMMA_BF16(xP, *(const v16bf*)(b2row[jj] + (NKT - 2) * 32), acc[jj]);
      }
      for (int jj = 0; jj < 4; ++jj) {
        acc[jj] = WMMA_BF16(hQ, *(const v16bf*)(b1row[jj] + (NKT - 1) * 32), acc[jj]);
        acc[jj] = WMMA_BF16(xQ, *(const v16bf*)(b2row[jj] + (NKT - 1) * 32), acc[jj]);
      }
    }
    for (int jj = 0; jj < 4; ++jj) {
      float vv = Ve[(ntg * 4 + jj) * 16 + ln];
      for (int r = 0; r < 8; ++r) ep[r] += tanhf(acc[jj][r]) * vv;
    }
  }
  for (int off = 1; off < 16; off <<= 1)
    for (int r = 0; r < 8; ++r) ep[r] += __shfl_xor(ep[r], off, 32);
  if (ln == 0)
    for (int r = 0; r < 8; ++r)
      e[(size_t)b * NT + tt * 16 + hi * 8 + r] = ep[r];
}

// ---------------------------------------------------------------------------
// K5: per-batch softmax, context, output head. One WG per batch row.
// ---------------------------------------------------------------------------
__global__ void __launch_bounds__(256) k_final(const __bf16* __restrict__ hs,
                                               const float* __restrict__ e,
                                               const float* __restrict__ Wout,
                                               const float* __restrict__ bout,
                                               float* __restrict__ out) {
  __shared__ float sm[NT];
  __shared__ float red[256];
  const int b = blockIdx.x, tid = threadIdx.x;

  float e0 = e[(size_t)b * NT + tid], e1 = e[(size_t)b * NT + tid + 256];
  red[tid] = fmaxf(e0, e1);
  __syncthreads();
  for (int s = 128; s > 0; s >>= 1) { if (tid < s) red[tid] = fmaxf(red[tid], red[tid + s]); __syncthreads(); }
  float mx = red[0];
  __syncthreads();
  float x0 = __expf(e0 - mx), x1 = __expf(e1 - mx);
  sm[tid] = x0; sm[tid + 256] = x1;
  red[tid] = x0 + x1;
  __syncthreads();
  for (int s = 128; s > 0; s >>= 1) { if (tid < s) red[tid] += red[tid + s]; __syncthreads(); }
  float tot = red[0];
  __syncthreads();

  float c1 = 0.f, c2 = 0.f;  // two h-columns per thread
  const __bf16* hb = hs + (size_t)b * NT * NH;
  for (int t = 0; t < NT; ++t) {
    float a = sm[t];
    c1 += a * bf2f(hb[(size_t)t * NH + tid]);
    c2 += a * bf2f(hb[(size_t)t * NH + tid + 256]);
  }
  float p0 = c1 * Wout[tid] + c2 * Wout[tid + 256];
  float p1 = c1 * Wout[NH + tid] + c2 * Wout[NH + tid + 256];
  red[tid] = p0;
  __syncthreads();
  for (int s = 128; s > 0; s >>= 1) { if (tid < s) red[tid] += red[tid + s]; __syncthreads(); }
  float mu = red[0] / tot + bout[0];
  __syncthreads();
  red[tid] = p1;
  __syncthreads();
  for (int s = 128; s > 0; s >>= 1) { if (tid < s) red[tid] += red[tid + s]; __syncthreads(); }
  if (tid == 0) {
    float sp = red[0] / tot + bout[1];
    out[b] = mu;
    float spl = sp > 20.f ? sp : log1pf(__expf(sp));
    out[NB + b] = spl + 1e-5f;
  }
}

// ---------------------------------------------------------------------------
extern "C" void kernel_launch(void* const* d_in, const int* in_sizes, int n_in,
                              void* d_out, int out_size, void* d_ws, size_t ws_size,
                              hipStream_t stream) {
  const float* x    = (const float*)d_in[0];
  const float* Wih  = (const float*)d_in[1];
  const float* Whh  = (const float*)d_in[2];
  const float* bih  = (const float*)d_in[3];
  const float* bhh  = (const float*)d_in[4];
  const float* Ve   = (const float*)d_in[5];
  const float* U1   = (const float*)d_in[6];
  const float* U2   = (const float*)d_in[7];
  const float* Wout = (const float*)d_in[8];
  const float* bout = (const float*)d_in[9];
  float* out = (float*)d_out;

  char* ws = (char*)d_ws;
  size_t o = 0;
  auto alloc = [&](size_t bytes) { size_t r = o; o += (bytes + 255) & ~(size_t)255; return r; };

  int*    bar    = (int*)   (ws + alloc(256));
  float*  bias   = (float*) (ws + alloc((size_t)NG * 4));
  __bf16* x_bf   = (__bf16*)(ws + alloc((size_t)NB * NT * NH * 2));
  __bf16* wih_bf = (__bf16*)(ws + alloc((size_t)NG * NH * 2));
  __bf16* whh_bf = (__bf16*)(ws + alloc((size_t)NG * NH * 2));
  __bf16* u1_bf  = (__bf16*)(ws + alloc((size_t)NH * 2 * NH * 2));
  __bf16* u2_bf  = (__bf16*)(ws + alloc((size_t)NH * NH * 2));
  __bf16* xg     = (__bf16*)(ws + alloc((size_t)NT * NB * NG * 2));   // 256 MB, tile order
  __bf16* hA     = (__bf16*)(ws + alloc((size_t)NB * NH * 2));
  __bf16* hB     = (__bf16*)(ws + alloc((size_t)NB * NH * 2));
  __bf16* hs     = (__bf16*)(ws + alloc((size_t)NB * NT * NH * 2));
  float*  rbuf   = (float*) (ws + alloc((size_t)NB * NH * 4));
  float*  ebuf   = (float*) (ws + alloc((size_t)NB * NT * 4));

  k_prep<<<2048, 256, 0, stream>>>(x, Wih, Whh, bih, bhh, U1, U2,
                                   x_bf, wih_bf, whh_bf, u1_bf, u2_bf, bias, hA, bar);
  k_xg<<<NT * 8 * 32 / 8, 256, 0, stream>>>(x_bf, wih_bf, bias, xg);     // 16384 blocks
  k_lstm<<<LSTM_WGS, 256, 0, stream>>>(xg, whh_bf, hA, hB, hs, bar);
  k_hlast<<<32, 256, 0, stream>>>(hs, u1_bf, rbuf);
  k_score<<<NB * 32 / 8, 256, 0, stream>>>(hs, x_bf, u1_bf, u2_bf, rbuf, Ve, ebuf); // 512 blocks
  k_final<<<NB, 256, 0, stream>>>(hs, ebuf, Wout, bout, out);
}
